// GraphTransformerLayer_45947560132964
// MI455X (gfx1250) — matrix operand order repaired and checked by reading, not compiled, not run on hardware
//
#include <hip/hip_runtime.h>
#include <hip/hip_bf16.h>

// ---------------------------------------------------------------------------
// GraphTransformerLayer for MI455X (gfx1250, wave32, WMMA).
// N=8192 atoms, M=16 bonds, H=256, NH=8 heads (head_dim=32).
// Compute-bound on the dense 8192x8192 attention (~69 GFLOP) -> all matmuls
// routed through v_wmma_f32_16x16x32_f16 (f32->f16 fragments, f32 accum).
// Row reductions use DPP16 VALU butterflies (no ds_bpermute); softmax row
// sums use a WMMA against an all-ones fragment.
// ---------------------------------------------------------------------------

typedef _Float16 v16h __attribute__((ext_vector_type(16)));
typedef _Float16 v8h  __attribute__((ext_vector_type(8)));
typedef _Float16 v2h  __attribute__((ext_vector_type(2)));
typedef float    v8f  __attribute__((ext_vector_type(8)));

#define WMMA_F32_F16(a, b, c) \
    __builtin_amdgcn_wmma_f32_16x16x32_f16(false, (a), false, (b), (short)0, (c), false, false)

static constexpr int NA   = 8192;   // num atoms
static constexpr int MB   = 16;     // max bonds
static constexpr int H    = 256;    // hidden
static constexpr int NH   = 8;      // heads
static constexpr int HD   = 32;     // head dim

// ---- DPP16 butterfly reductions within each 16-lane row (wave32) ----------
// Order matters: xor1, xor2 make quads uniform, then row_half_mirror acts as
// xor4 and row_mirror acts as xor8.
template<int CTRL>
__device__ __forceinline__ float dpp_max_step(float v) {
    int a = __float_as_int(v);
    int t = __builtin_amdgcn_update_dpp(a, a, CTRL, 0xF, 0xF, true);
    return fmaxf(v, __int_as_float(t));
}
template<int CTRL>
__device__ __forceinline__ float dpp_add_step(float v) {
    int a = __float_as_int(v);
    int t = __builtin_amdgcn_update_dpp(a, a, CTRL, 0xF, 0xF, true);
    return v + __int_as_float(t);
}
__device__ __forceinline__ float row16_max(float v) {
    v = dpp_max_step<0xB1>(v);    // quad_perm [1,0,3,2]  (xor 1)
    v = dpp_max_step<0x4E>(v);    // quad_perm [2,3,0,1]  (xor 2)
    v = dpp_max_step<0x141>(v);   // row_half_mirror      (xor 4 eff.)
    v = dpp_max_step<0x140>(v);   // row_mirror           (xor 8 eff.)
    return v;
}
__device__ __forceinline__ float row16_sum(float v) {
    v = dpp_add_step<0xB1>(v);
    v = dpp_add_step<0x4E>(v);
    v = dpp_add_step<0x141>(v);
    v = dpp_add_step<0x140>(v);
    return v;
}

// ===========================================================================
// Generic WMMA GEMM:  C[M,N] = epilogue( A[M,K] @ W[K,N] + bias )
//   CONCAT: A element (row,k): k <  Ksplit -> A1[(row>>rowShift)*ldA1 + k]
//                              k >= Ksplit -> A2[row*ldA2 + (k-Ksplit)]
//   (Ksplit % 32 == 0, so the A1/A2 choice is uniform per 32-wide K step.)
// Block: 256 threads = 8 waves, macro tile 64(M) x 64(N); each wave owns a
// 16x32 strip (1 A frag, 2 B frags, 2 WMMAs per K step of 32).
// LDS staged as packed f16 pairs (ds_store_b32). Requires M%64==0, N%64==0,
// K%32==0 (true for every call here).
// ===========================================================================
template<bool CONCAT, bool RELU, bool HASRES>
__global__ __launch_bounds__(256)
void wmma_gemm_kernel(const float* __restrict__ A1, int ldA1, int rowShift,
                      const float* __restrict__ A2, int ldA2, int Ksplit,
                      const float* __restrict__ W,  const float* __restrict__ bias,
                      const float* __restrict__ residual,
                      float* __restrict__ C,
                      int Ntot, int Ktot)
{
    __shared__ _Float16 As[64 * 32];   // [row][k]
    __shared__ _Float16 BsT[64 * 32];  // [n][k]  (W tile transposed)

    const int tid  = threadIdx.x;
    const int lane = tid & 31;
    const int wave = tid >> 5;        // 0..7
    const int wm   = wave & 3;        // 0..3 -> 16-row sub-tile
    const int wn   = wave >> 2;       // 0..1 -> 32-col strip
    const int n16  = lane & 15;
    const int half = lane >> 4;

    const int rowBase = blockIdx.x * 64;
    const int colBase = blockIdx.y * 64;

    v8f acc0{}, acc1{};

    for (int kb = 0; kb < Ktot; kb += 32) {
        // prefetch next W tile into cache (global_prefetch_b8)
        if (kb + 32 < Ktot)
            __builtin_prefetch(&W[(size_t)(kb + 32 + (tid >> 6)) * Ntot + colBase + (tid & 63)], 0, 0);

        // ---- uniform A-source select for this K step ----
        const bool useA1 = (!CONCAT) || (kb < Ksplit);
        const float* __restrict__ base = useA1 ? A1 : A2;
        const int ld   = useA1 ? ldA1 : ldA2;
        const int koff = useA1 ? kb : (kb - Ksplit);
        const int rsh  = useA1 ? rowShift : 0;

        // ---- stage A tile: 64x32, packed f16 pairs, coalesced float2 reads
        #pragma unroll
        for (int i = 0; i < 4; ++i) {
            int p  = tid + i * 256;          // 1024 pairs
            int r  = p >> 4;                 // 64 rows
            int kp = p & 15;                 // 16 pairs / row
            int grow = rowBase + r;
            float2 f2 = *(const float2*)&base[(size_t)(grow >> rsh) * ld + koff + 2 * kp];
            v2h hp; hp[0] = (_Float16)f2.x; hp[1] = (_Float16)f2.y;
            *(v2h*)&As[r * 32 + 2 * kp] = hp;
        }
        // ---- stage W tile transposed: 32k x 64n, packed pairs along k ----
        #pragma unroll
        for (int i = 0; i < 4; ++i) {
            int p  = tid + i * 256;          // 1024 pairs
            int n  = p & 63;                 // coalesced along n
            int k  = (p >> 6) * 2;
            float b0 = W[(size_t)(kb + k) * Ntot + colBase + n];
            float b1 = W[(size_t)(kb + k + 1) * Ntot + colBase + n];
            v2h hp; hp[0] = (_Float16)b0; hp[1] = (_Float16)b1;
            *(v2h*)&BsT[n * 32 + k] = hp;
        }
        __syncthreads();

        // ---- fragments + WMMA ----
        // A frag: lane holds row m = lane%16; k = 16h+e contiguous
        v8h a0 = *(const v8h*)&As[(wm * 16 + n16) * 32 + half * 16];
        v8h a1 = *(const v8h*)&As[(wm * 16 + n16) * 32 + half * 16 + 8];
        v16h av = __builtin_shufflevector(a0, a1, 0,1,2,3,4,5,6,7,8,9,10,11,12,13,14,15);
        // B frags: lane holds col n = lane%16; k = 16h+e contiguous
        v16h bv0 = *(const v16h*)&BsT[(wn * 32 + n16) * 32 + half * 16];
        v16h bv1 = *(const v16h*)&BsT[(wn * 32 + 16 + n16) * 32 + half * 16];

        acc0 = WMMA_F32_F16(av, bv0, acc0);
        acc1 = WMMA_F32_F16(av, bv1, acc1);
        __syncthreads();
    }

    // ---- epilogue: C/D layout lane holds D[r + 8*half][lane%16] ----
    const int col0 = colBase + wn * 32 + n16;
    const int col1 = col0 + 16;
    const float b0 = bias[col0];
    const float b1 = bias[col1];
    #pragma unroll
    for (int r = 0; r < 8; ++r) {
        int row = rowBase + wm * 16 + r + half * 8;
        float v0 = acc0[r] + b0;
        float v1 = acc1[r] + b1;
        if (RELU) { v0 = fmaxf(v0, 0.0f); v1 = fmaxf(v1, 0.0f); }
        if (HASRES) {
            v0 += residual[(size_t)row * Ntot + col0];
            v1 += residual[(size_t)row * Ntot + col1];
        }
        C[(size_t)row * Ntot + col0] = v0;
        C[(size_t)row * Ntot + col1] = v1;
    }
}

// ===========================================================================
// Flash attention. Block = 4 waves = 64 queries (16 per wave), one head per
// blockIdx.y. K tile (32 keys) and V^T tile staged cooperatively in LDS and
// shared by all 4 waves; per-wave online softmax (row max via DPP butterfly,
// row sum via WMMA against all-ones B); P round-trips through a private LDS
// strip to become an A fragment. head_dim = 32 = one WMMA K-step.
// qkv row layout: [q(256) | k(256) | v(256)].
// ===========================================================================
__global__ __launch_bounds__(128)
void flash_attn_kernel(const float* __restrict__ qkv, float* __restrict__ out, int N)
{
    __shared__ _Float16 Ks[32 * 32];       // [key][dim]
    __shared__ _Float16 VT[32 * 32];       // [dim][key]
    __shared__ _Float16 Plds[4][16 * 32];  // per-wave [query m][key col]

    const int tid  = threadIdx.x;
    const int lane = tid & 31;
    const int wv   = tid >> 5;            // 0..3
    const int n16  = lane & 15;
    const int half = lane >> 4;
    const int qBase = (blockIdx.x * 4 + wv) * 16;
    const int head  = blockIdx.y;
    const float qscale = 0.17677669529663687f;   // 1/sqrt(32)

    // Q fragment (A layout), scale folded into the f16 conversion
    v16h aq;
    {
        const float* qrow = qkv + (size_t)(qBase + n16) * 768 + head * HD;
        #pragma unroll
        for (int i = 0; i < 8; ++i) {
            aq[i]     = (_Float16)(qrow[half * 16 + i] * qscale);
            aq[8 + i] = (_Float16)(qrow[half * 16 + 8 + i] * qscale);
        }
    }
    // all-ones B fragment: rowsum(P) = P @ ones lands per-lane in D layout
    v16h vones;
    #pragma unroll
    for (int e = 0; e < 16; ++e) vones[e] = (_Float16)1.0f;

    v8f o0{}, o1{};
    float mx[8], li[8];
    #pragma unroll
    for (int r = 0; r < 8; ++r) { mx[r] = -1e30f; li[r] = 0.0f; }

    for (int kb = 0; kb < N; kb += 32) {
        // ---- cooperative staging (128 threads, packed f16 pairs) ----
        // K tile, row-major [key][dim] (B-frag wants contiguous dims)
        #pragma unroll
        for (int i = 0; i < 4; ++i) {
            int p   = tid + i * 128;       // 512 float2
            int key = p >> 4;
            int dp  = p & 15;              // dim pair
            float2 kf = *(const float2*)&qkv[(size_t)(kb + key) * 768 + H + head * HD + 2 * dp];
            v2h kh; kh[0] = (_Float16)kf.x; kh[1] = (_Float16)kf.y;
            *(v2h*)&Ks[key * 32 + 2 * dp] = kh;
        }
        // V tile transposed [dim][key]: 2x2 (key-pair x dim-pair) patches so
        // both LDS stores are packed 4-byte writes along the key dimension
        #pragma unroll
        for (int i = 0; i < 2; ++i) {
            int p  = tid + i * 128;        // 256 patches
            int kp = p >> 4;               // key pair
            int dp = p & 15;               // dim pair
            float2 va = *(const float2*)&qkv[(size_t)(kb + 2 * kp)     * 768 + 2 * H + head * HD + 2 * dp];
            float2 vb = *(const float2*)&qkv[(size_t)(kb + 2 * kp + 1) * 768 + 2 * H + head * HD + 2 * dp];
            v2h r0; r0[0] = (_Float16)va.x; r0[1] = (_Float16)vb.x;   // dim 2dp
            v2h r1; r1[0] = (_Float16)va.y; r1[1] = (_Float16)vb.y;   // dim 2dp+1
            *(v2h*)&VT[(2 * dp)     * 32 + 2 * kp] = r0;
            *(v2h*)&VT[(2 * dp + 1) * 32 + 2 * kp] = r1;
        }
        __syncthreads();

        // ---- S = Q K^T : two 16-key column tiles ----
        v16h bk0 = *(const v16h*)&Ks[n16 * 32 + half * 16];
        v16h bk1 = *(const v16h*)&Ks[(16 + n16) * 32 + half * 16];
        v8f z{};
        v8f s0 = WMMA_F32_F16(aq, bk0, z);
        v8f s1 = WMMA_F32_F16(aq, bk1, z);

        // ---- online softmax: row max via DPP (row m = r + 8*half) ----
        float lcorr[8];
        #pragma unroll
        for (int r = 0; r < 8; ++r) {
            float m = row16_max(fmaxf(s0[r], s1[r]));
            float newm = fmaxf(mx[r], m);
            float p0 = __expf(s0[r] - newm);
            float p1 = __expf(s1[r] - newm);
            float corr = __expf(mx[r] - newm);
            lcorr[r] = corr;
            mx[r] = newm;
            o0[r] *= corr;
            o1[r] *= corr;
            Plds[wv][(r + half * 8) * 32 + n16]      = (_Float16)p0;
            Plds[wv][(r + half * 8) * 32 + 16 + n16] = (_Float16)p1;
        }

        // ---- P fragment (A layout) from private LDS strip ----
        v8h pa = *(const v8h*)&Plds[wv][n16 * 32 + half * 16];
        v8h pb = *(const v8h*)&Plds[wv][n16 * 32 + half * 16 + 8];
        v16h ap = __builtin_shufflevector(pa, pb, 0,1,2,3,4,5,6,7,8,9,10,11,12,13,14,15);

        // ---- row sums of P via WMMA against ones (replaces 32 bpermutes) ----
        v8f rs = WMMA_F32_F16(ap, vones, z);
        #pragma unroll
        for (int r = 0; r < 8; ++r) li[r] = li[r] * lcorr[r] + rs[r];

        // ---- O += P V  (V^T shared) ----
        v16h bv0 = *(const v16h*)&VT[n16 * 32 + half * 16];
        v16h bv1 = *(const v16h*)&VT[(16 + n16) * 32 + half * 16];
        o0 = WMMA_F32_F16(ap, bv0, o0);
        o1 = WMMA_F32_F16(ap, bv1, o1);
        __syncthreads();
    }

    // ---- normalize + write [N, 256] at this head's 32 dims ----
    #pragma unroll
    for (int r = 0; r < 8; ++r) {
        float inv = 1.0f / li[r];
        int row = qBase + r + half * 8;
        out[(size_t)row * H + head * HD + n16]      = o0[r] * inv;
        out[(size_t)row * H + head * HD + 16 + n16] = o1[r] * inv;
    }
}

// ===========================================================================
// LayerNorm over last dim (256). One wave per row, 8 rows per block.
// ===========================================================================
__global__ __launch_bounds__(256)
void ln_kernel(const float* __restrict__ in, const float* __restrict__ scale,
               const float* __restrict__ bias, float* __restrict__ out, int rows)
{
    const int wv = threadIdx.x >> 5, lane = threadIdx.x & 31;
    const int row = blockIdx.x * 8 + wv;
    if (row >= rows) return;
    const float* p = in + (size_t)row * H;
    float v[8];
    float s = 0.0f;
    #pragma unroll
    for (int i = 0; i < 8; ++i) { v[i] = p[lane + i * 32]; s += v[i]; }
    s = row16_sum(s);
    s += __shfl_xor(s, 16, 32);
    const float mu = s * (1.0f / H);
    float q = 0.0f;
    #pragma unroll
    for (int i = 0; i < 8; ++i) { float d = v[i] - mu; q += d * d; }
    q = row16_sum(q);
    q += __shfl_xor(q, 16, 32);
    const float rs = rsqrtf(q * (1.0f / H) + 1e-5f);
    #pragma unroll
    for (int i = 0; i < 8; ++i) {
        int c = lane + i * 32;
        out[(size_t)row * H + c] = (v[i] - mu) * rs * scale[c] + bias[c];
    }
}

// ===========================================================================
// neighbor scores = h[131072,256] @ na_w2[256,8] + b2  (tiny N -> VALU dot)
// one thread per (row, head)
// ===========================================================================
__global__ __launch_bounds__(256)
void na2_kernel(const float* __restrict__ h, const float* __restrict__ w2,
                const float* __restrict__ b2, float* __restrict__ scores, int rows)
{
    const int p = blockIdx.x * 256 + threadIdx.x;
    if (p >= rows * NH) return;
    const int row = p >> 3, head = p & 7;
    const float* hp = h + (size_t)row * H;
    float acc = b2[head];
    #pragma unroll 4
    for (int c = 0; c < H; ++c) acc += hp[c] * w2[c * NH + head];
    scores[p] = acc;
}

// ===========================================================================
// softmax over bonds, mean over heads, weighted neighbor message, residual.
// One wave per atom (8 atoms per 256-thread block).
// ===========================================================================
__global__ __launch_bounds__(256)
void neighbor_msg_kernel(const float* __restrict__ scores, const float* __restrict__ x1,
                         const float* __restrict__ neighbors, const float* __restrict__ edge,
                         float* __restrict__ x2, int natoms)
{
    __shared__ float wgt[8][16];
    const int wv = threadIdx.x >> 5, lane = threadIdx.x & 31;
    const int atom = blockIdx.x * 8 + wv;
    if (atom >= natoms) return;
    const int bond = lane & 15;

    const float* sp = scores + (size_t)atom * MB * NH;
    float wsum = 0.0f;
    #pragma unroll
    for (int head = 0; head < NH; ++head) {
        float v = sp[bond * NH + head];
        float m = row16_max(v);
        float e = __expf(v - m);
        float s = row16_sum(e);
        wsum += e / s;
    }
    if (lane < 16) wgt[wv][bond] = wsum * (1.0f / NH);
    __syncthreads();

    const float* np = neighbors + (size_t)atom * MB * H;
    const float* ep = edge      + (size_t)atom * MB * H;
    for (int c = lane; c < H; c += 32) {
        float msg = 0.0f;
        #pragma unroll
        for (int b = 0; b < MB; ++b)
            msg += np[b * H + c] * wgt[wv][b] * ep[b * H + c];
        x2[(size_t)atom * H + c] = x1[(size_t)atom * H + c] + msg;
    }
}

// ===========================================================================
// host launcher
// ===========================================================================
extern "C" void kernel_launch(void* const* d_in, const int* in_sizes, int n_in,
                              void* d_out, int out_size, void* d_ws, size_t ws_size,
                              hipStream_t stream)
{
    const float* x         = (const float*)d_in[0];
    const float* neighbors = (const float*)d_in[1];
    const float* edge      = (const float*)d_in[2];
    const float* in_w      = (const float*)d_in[3];
    const float* in_b      = (const float*)d_in[4];
    const float* out_w     = (const float*)d_in[5];
    const float* out_b     = (const float*)d_in[6];
    const float* ln1_s     = (const float*)d_in[7];
    const float* ln1_b     = (const float*)d_in[8];
    const float* na_w1     = (const float*)d_in[9];
    const float* na_b1     = (const float*)d_in[10];
    const float* na_w2     = (const float*)d_in[11];
    const float* na_b2     = (const float*)d_in[12];
    const float* ff_w1     = (const float*)d_in[13];
    const float* ff_b1     = (const float*)d_in[14];
    const float* ff_w2     = (const float*)d_in[15];
    const float* ff_b2     = (const float*)d_in[16];
    const float* ln2_s     = (const float*)d_in[17];
    const float* ln2_b     = (const float*)d_in[18];
    float* outp = (float*)d_out;

    // workspace regions (floats), aliased by lifetime:
    //  A (33.55M f): qkv -> h_na -> ff1      B (2.10M f): attn_out -> scores
    //  C (2.10M f): res1 -> res2             D: x1          E: x2
    const size_t SZ_X  = (size_t)NA * H;          // 2,097,152
    const size_t REG_A = (size_t)NA * MB * H;     // 33,554,432
    float* ws = (float*)d_ws;
    float* qkv      = ws;                 // [8192, 768]
    float* h_na     = ws;                 // [131072, 256]
    float* ff1      = ws;                 // [8192, 1024]
    float* attn_out = ws + REG_A;         // [8192, 256]
    float* scoresB  = ws + REG_A;         // [131072, 8]
    float* res1     = ws + REG_A + SZ_X;  // pre-LN1
    float* res2     = ws + REG_A + SZ_X;  // pre-LN2
    float* x1       = ws + REG_A + 2 * SZ_X;
    float* x2       = ws + REG_A + 3 * SZ_X;

    dim3 blk256(256), blk128(128);

    // 1) qkv = x @ in_proj_w + b             [8192,768]
    wmma_gemm_kernel<false, false, false><<<dim3(NA / 64, (3 * H) / 64), blk256, 0, stream>>>(
        x, H, 0, x, H, H, in_w, in_b, nullptr, qkv, 3 * H, H);

    // 2) flash attention -> attn_out         [8192,256]
    flash_attn_kernel<<<dim3(NA / 64, NH), blk128, 0, stream>>>(qkv, attn_out, NA);

    // 3) res1 = attn_out @ out_proj_w + b + x
    wmma_gemm_kernel<false, false, true><<<dim3(NA / 64, H / 64), blk256, 0, stream>>>(
        attn_out, H, 0, attn_out, H, H, out_w, out_b, x, res1, H, H);

    // 4) x1 = LN1(res1)
    ln_kernel<<<dim3(NA / 8), blk256, 0, stream>>>(res1, ln1_s, ln1_b, x1, NA);

    // 5) h_na = relu([x1_exp | neighbors] @ na_w1 + b1)   [131072,256], K=512
    wmma_gemm_kernel<true, true, false><<<dim3((NA * MB) / 64, H / 64), blk256, 0, stream>>>(
        x1, H, 4, neighbors, H, H, na_w1, na_b1, nullptr, h_na, H, 2 * H);

    // 6) scores = h_na @ na_w2 + b2          [131072,8]
    na2_kernel<<<dim3((NA * MB * NH) / 256), blk256, 0, stream>>>(
        h_na, na_w2, na_b2, scoresB, NA * MB);

    // 7) x2 = x1 + weighted neighbor message
    neighbor_msg_kernel<<<dim3(NA / 8), blk256, 0, stream>>>(
        scoresB, x1, neighbors, edge, x2, NA);

    // 8) ff1 = relu(x2 @ ff_w1 + b1)         [8192,1024]
    wmma_gemm_kernel<false, true, false><<<dim3(NA / 64, (4 * H) / 64), blk256, 0, stream>>>(
        x2, H, 0, x2, H, H, ff_w1, ff_b1, nullptr, ff1, 4 * H, H);

    // 9) res2 = ff1 @ ff_w2 + b2 + x2
    wmma_gemm_kernel<false, false, true><<<dim3(NA / 64, H / 64), blk256, 0, stream>>>(
        ff1, 4 * H, 0, ff1, 4 * H, 4 * H, ff_w2, ff_b2, x2, res2, H, 4 * H);

    // 10) out = LN2(res2)
    ln_kernel<<<dim3(NA / 8), blk256, 0, stream>>>(res2, ln2_s, ln2_b, outp, NA);
}
